// DiTAttention_37151467110710
// MI455X (gfx1250) — matrix operand order repaired and checked by reading, not compiled, not run on hardware
//
#include <hip/hip_runtime.h>

// Problem constants (match reference)
#define B_  8
#define S_  1024
#define D_  1024
#define H_  16
#define HD_ 64
#define M_  (B_ * S_)   // 8192 rows for projection GEMMs

typedef __attribute__((ext_vector_type(16))) __bf16 v16bf;
typedef __attribute__((ext_vector_type(8)))  __bf16 v8bf;
typedef __attribute__((ext_vector_type(8)))  float  v8f;

union V16U { v16bf v; v8bf h[2]; };
union V8SU { v8bf v; unsigned short s[8]; };

__device__ __forceinline__ unsigned short f32_to_bf16(float f) {
  unsigned int u = __float_as_uint(f);
  return (unsigned short)((u + (0x7FFFu + ((u >> 16) & 1u))) >> 16); // RNE
}

#define WMMA_BF16(a, b, c)                                                     \
  __builtin_amdgcn_wmma_f32_16x16x32_bf16(false, (a), false, (b), (short)0,    \
                                          (c), false, false)

// Async global->LDS copy, 16B per lane, tracked by ASYNCcnt
// (CDNA5 ISA 15.18.3 op 98: GLOBAL_LOAD_ASYNC_TO_LDS_B128; VDST = LDS addr).
__device__ __forceinline__ void async_copy_b128(const unsigned short* g,
                                                unsigned short* l) {
  unsigned lds = (unsigned)(unsigned long long)l;  // low 32 bits = LDS offset
  asm volatile("global_load_async_to_lds_b128 %0, %1, off"
               :: "v"(lds), "v"(g) : "memory");
}
__device__ __forceinline__ void wait_asynccnt0() {
  asm volatile("s_wait_asynccnt 0x0" ::: "memory");
}

// ---------------------------------------------------------------------------
// fp32 -> bf16 conversion (grid-stride)
// ---------------------------------------------------------------------------
__global__ void convert_f32_to_bf16(const float* __restrict__ in,
                                    unsigned short* __restrict__ out, int n) {
  int i = blockIdx.x * blockDim.x + threadIdx.x;
  int stride = gridDim.x * blockDim.x;
  for (; i < n; i += stride) out[i] = f32_to_bf16(in[i]);
}

// ---------------------------------------------------------------------------
// GEMM: C[M,N] = A[M,K](bf16) * W[N,K]^T (bf16), fp32 accumulate.
// Block 256 = 8 waves; wave tile 32Mx32N; block tile 128Mx64N.
// Double-buffered LDS tiles filled with async global->LDS copies:
//   ldsA: 128x32 bf16 (8KB)  -- 512 x b128 chunks, 2 per thread
//   ldsB:  64x32 bf16 (4KB)  -- 256 x b128 chunks, 1 per thread
// Fragment layouts (ISA 7.12.2):
//   A-frag: lane L row (L&15); halves [k+(L>>4)*8 ..+7] and [+16 ..+7]
//   B-frag: lane L col (L&15); halves [k+(L>>4)*8 ..+7] and [+16 ..+7]
//   C/D:    vgpr j, lane L -> row j+8*(L>>4), col (L&15)
// ---------------------------------------------------------------------------
template <bool OUT_BF16>
__global__ __launch_bounds__(256) void gemm_bf16_wmma(
    const unsigned short* __restrict__ A,   // [M,K] bf16
    const unsigned short* __restrict__ W,   // [N,K] bf16 (nn.Linear weight)
    void* __restrict__ Cout,                // [M,N] bf16 or f32
    int M, int N, int K) {
  __shared__ unsigned short ldsA[2][128 * 32];
  __shared__ unsigned short ldsB[2][64 * 32];

  const int tid  = threadIdx.x;
  const int lane = tid & 31;
  const int wave = tid >> 5;
  const int hl   = lane >> 4;   // 0/1 half of wave
  const int l16  = lane & 15;

  const int mBlock = blockIdx.x * 128;
  const int nBlock = blockIdx.y * 64;
  const int mo = (wave & 3) * 32;   // wave M offset within block tile
  const int no = (wave >> 2) * 32;  // wave N offset within block tile

  // Per-thread async-copy assignments (8 halves = 16B per chunk)
  const int coff  = (tid & 3) * 8;  // 0,8,16,24
  const int arow0 = tid >> 2;       // 0..63
  const int arow1 = arow0 + 64;     // 64..127
  const int brow  = tid >> 2;       // 0..63

  const unsigned short* gA0 = A + (size_t)(mBlock + arow0) * K + coff;
  const unsigned short* gA1 = A + (size_t)(mBlock + arow1) * K + coff;
  const unsigned short* gB  = W + (size_t)(nBlock + brow) * K + coff;

  v8f c00 = {}, c01 = {}, c10 = {}, c11 = {};

  // prologue prefetch
  async_copy_b128(gA0, &ldsA[0][arow0 * 32 + coff]);
  async_copy_b128(gA1, &ldsA[0][arow1 * 32 + coff]);
  async_copy_b128(gB,  &ldsB[0][brow * 32 + coff]);

  const int steps = K / 32;
  for (int i = 0; i < steps; ++i) {
    const int buf = i & 1;
    wait_asynccnt0();    // this wave's copies into buf have landed in LDS
    __syncthreads();     // all waves' copies landed; prev compute reads done

    if (i + 1 < steps) {
      const int k = (i + 1) * 32;
      async_copy_b128(gA0 + k, &ldsA[buf ^ 1][arow0 * 32 + coff]);
      async_copy_b128(gA1 + k, &ldsA[buf ^ 1][arow1 * 32 + coff]);
      async_copy_b128(gB + k,  &ldsB[buf ^ 1][brow * 32 + coff]);
    }

    v16bf a0, a1, b0, b1;
    {
      V16U u;
      const unsigned short* pa = &ldsA[buf][(mo + l16) * 32 + hl * 8];
      u.h[0] = *(const v8bf*)pa;
      u.h[1] = *(const v8bf*)(pa + 16);
      a0 = u.v;
      const unsigned short* pa2 = pa + 16 * 32;
      u.h[0] = *(const v8bf*)pa2;
      u.h[1] = *(const v8bf*)(pa2 + 16);
      a1 = u.v;
      const unsigned short* pb = &ldsB[buf][(no + l16) * 32 + hl * 8];
      u.h[0] = *(const v8bf*)pb;
      u.h[1] = *(const v8bf*)(pb + 16);
      b0 = u.v;
      const unsigned short* pb2 = pb + 16 * 32;
      u.h[0] = *(const v8bf*)pb2;
      u.h[1] = *(const v8bf*)(pb2 + 16);
      b1 = u.v;
    }
    c00 = WMMA_BF16(a0, b0, c00);
    c01 = WMMA_BF16(a0, b1, c01);
    c10 = WMMA_BF16(a1, b0, c10);
    c11 = WMMA_BF16(a1, b1, c11);
  }

#pragma unroll
  for (int j = 0; j < 8; ++j) {
    const int r0 = mBlock + mo + j + 8 * hl;
    const int r1 = r0 + 16;
    const int ca = nBlock + no + l16;
    const int cb = ca + 16;
    if (OUT_BF16) {
      unsigned short* C = (unsigned short*)Cout;
      C[(size_t)r0 * N + ca] = f32_to_bf16(c00[j]);
      C[(size_t)r0 * N + cb] = f32_to_bf16(c01[j]);
      C[(size_t)r1 * N + ca] = f32_to_bf16(c10[j]);
      C[(size_t)r1 * N + cb] = f32_to_bf16(c11[j]);
    } else {
      float* C = (float*)Cout;
      C[(size_t)r0 * N + ca] = c00[j];
      C[(size_t)r0 * N + cb] = c01[j];
      C[(size_t)r1 * N + ca] = c10[j];
      C[(size_t)r1 * N + cb] = c11[j];
    }
  }
}

// ---------------------------------------------------------------------------
// Flash attention per (b,h). grid = (B*H, S/128), block = 256 (8 waves).
// Each wave owns 16 query rows; workgroup stages K (async, row-major) and V^T
// in LDS per 32-key block; online softmax in f32; P re-laid via per-wave LDS.
// ---------------------------------------------------------------------------
__global__ __launch_bounds__(256) void flash_attn_bf16(
    const unsigned short* __restrict__ Q,   // [B*S, D] bf16
    const unsigned short* __restrict__ K,
    const unsigned short* __restrict__ V,
    unsigned short* __restrict__ O) {       // [B*S, D] bf16 context
  const int bh = blockIdx.x;
  const int b  = bh / H_;
  const int h  = bh % H_;
  const int tid  = threadIdx.x;
  const int lane = tid & 31;
  const int wave = tid >> 5;
  const int hl   = lane >> 4;
  const int l16  = lane & 15;

  const size_t baseRow = (size_t)b * S_;
  const unsigned short* Qb = Q + baseRow * D_ + h * HD_;
  const unsigned short* Kb = K + baseRow * D_ + h * HD_;
  const unsigned short* Vb = V + baseRow * D_ + h * HD_;
  unsigned short*       Ob = O + baseRow * D_ + h * HD_;

  __shared__ unsigned short ldsK[32 * 64];     // [key][hd]
  __shared__ unsigned short ldsVT[64 * 32];    // [hd][key]
  __shared__ unsigned short ldsP[8][16 * 32];  // per-wave P staging [row][key]

  const int qBase = blockIdx.y * 128 + wave * 16;  // query row in 0..1023

  // Load Q fragments for this wave's 16 rows (hd split into two k=32 frags)
  v16bf qf0, qf1;
  {
    const unsigned short* qr = Qb + (size_t)(qBase + l16) * D_;
    V16U u;
    u.h[0] = *(const v8bf*)(qr + hl * 8);
    u.h[1] = *(const v8bf*)(qr + hl * 8 + 16);
    qf0 = u.v;
    u.h[0] = *(const v8bf*)(qr + 32 + hl * 8);
    u.h[1] = *(const v8bf*)(qr + 32 + hl * 8 + 16);
    qf1 = u.v;
  }

  v8f o0 = {}, o1 = {}, o2 = {}, o3 = {};
  float mrow[8], lrow[8];
#pragma unroll
  for (int j = 0; j < 8; ++j) { mrow[j] = -__builtin_inff(); lrow[j] = 0.f; }

  const float scale = 0.125f;  // 1/sqrt(64)

  for (int kb = 0; kb < S_; kb += 32) {
    __syncthreads();  // protect LDS from previous iteration readers
    // Stage K block rows kb..kb+31 into ldsK (row-major) with async copies;
    // stage V transposed into ldsVT[hd][key] through registers.
    {
      const int r = tid >> 3;          // 0..31
      const int c = (tid & 7) * 8;     // 0..56 step 8
      async_copy_b128(Kb + (size_t)(kb + r) * D_ + c, &ldsK[r * 64 + c]);
      V8SU t;
      t.v = *(const v8bf*)(Vb + (size_t)(kb + r) * D_ + c);
#pragma unroll
      for (int i = 0; i < 8; ++i) ldsVT[(c + i) * 32 + r] = t.s[i];
    }
    wait_asynccnt0();
    __syncthreads();

    // s0: keys kb..kb+15 ; s1: keys kb+16..kb+31  (QK^T, k = hd)
    v8f s0 = {}, s1 = {};
#pragma unroll
    for (int c = 0; c < 64; c += 32) {
      V16U ub;
      const unsigned short* kp0 = &ldsK[(0 + l16) * 64 + c + hl * 8];
      ub.h[0] = *(const v8bf*)kp0;
      ub.h[1] = *(const v8bf*)(kp0 + 16);
      s0 = WMMA_BF16((c == 0 ? qf0 : qf1), ub.v, s0);
      const unsigned short* kp1 = &ldsK[(16 + l16) * 64 + c + hl * 8];
      ub.h[0] = *(const v8bf*)kp1;
      ub.h[1] = *(const v8bf*)(kp1 + 16);
      s1 = WMMA_BF16((c == 0 ? qf0 : qf1), ub.v, s1);
    }

    // Online softmax update. Row of (s*,j) = j + 8*hl; cols striped over 16
    // lanes of each half -> reduce with xor masks 1,2,4,8.
    float p0[8], p1[8];
#pragma unroll
    for (int j = 0; j < 8; ++j) {
      float a0 = s0[j] * scale;
      float a1 = s1[j] * scale;
      float bmax = fmaxf(a0, a1);
#pragma unroll
      for (int mk = 1; mk < 16; mk <<= 1)
        bmax = fmaxf(bmax, __shfl_xor(bmax, mk, 32));
      const float mnew = fmaxf(mrow[j], bmax);
      const float corr = __expf(mrow[j] - mnew);
      const float e0 = __expf(a0 - mnew);
      const float e1 = __expf(a1 - mnew);
      float rs = e0 + e1;
#pragma unroll
      for (int mk = 1; mk < 16; mk <<= 1) rs += __shfl_xor(rs, mk, 32);
      lrow[j] = lrow[j] * corr + rs;
      mrow[j] = mnew;
      o0[j] *= corr; o1[j] *= corr; o2[j] *= corr; o3[j] *= corr;
      p0[j] = e0;
      p1[j] = e1;
    }

    // Re-lay P (C layout) into A layout via per-wave LDS staging.
    {
      unsigned short* Pw = ldsP[wave];
#pragma unroll
      for (int j = 0; j < 8; ++j) {
        const int r = j + 8 * hl;
        Pw[r * 32 + l16]      = f32_to_bf16(p0[j]);
        Pw[r * 32 + 16 + l16] = f32_to_bf16(p1[j]);
      }
    }
    v16bf pf;
    {
      const unsigned short* pr = ldsP[wave] + l16 * 32 + hl * 8;
      V16U u;
      u.h[0] = *(const v8bf*)pr;
      u.h[1] = *(const v8bf*)(pr + 16);
      pf = u.v;
    }

    // O += P * V  (B-frag: col = hd, k = key; from ldsVT rows, contiguous)
#pragma unroll
    for (int t = 0; t < 4; ++t) {
      V16U uv;
      const unsigned short* vp = &ldsVT[(t * 16 + l16) * 32 + hl * 8];
      uv.h[0] = *(const v8bf*)vp;
      uv.h[1] = *(const v8bf*)(vp + 16);
      if (t == 0) o0 = WMMA_BF16(pf, uv.v, o0);
      if (t == 1) o1 = WMMA_BF16(pf, uv.v, o1);
      if (t == 2) o2 = WMMA_BF16(pf, uv.v, o2);
      if (t == 3) o3 = WMMA_BF16(pf, uv.v, o3);
    }
  }

  // Normalize and store context (bf16) back into [B*S, D] layout.
#pragma unroll
  for (int j = 0; j < 8; ++j) {
    const int row = qBase + j + 8 * hl;
    const float inv = 1.0f / lrow[j];
    unsigned short* orow = Ob + (size_t)row * D_;
    orow[0  + l16] = f32_to_bf16(o0[j] * inv);
    orow[16 + l16] = f32_to_bf16(o1[j] * inv);
    orow[32 + l16] = f32_to_bf16(o2[j] * inv);
    orow[48 + l16] = f32_to_bf16(o3[j] * inv);
  }
}

// ---------------------------------------------------------------------------
// Launch
// ---------------------------------------------------------------------------
extern "C" void kernel_launch(void* const* d_in, const int* in_sizes, int n_in,
                              void* d_out, int out_size, void* d_ws,
                              size_t ws_size, hipStream_t stream) {
  (void)in_sizes; (void)n_in; (void)out_size; (void)ws_size;
  const float* x  = (const float*)d_in[0];
  const float* wq = (const float*)d_in[1];
  const float* wk = (const float*)d_in[2];
  const float* wv = (const float*)d_in[3];
  const float* wo = (const float*)d_in[4];

  const int nX = M_ * D_;   // 8388608
  const int nW = D_ * D_;   // 1048576

  unsigned short* ws  = (unsigned short*)d_ws;
  unsigned short* xb  = ws;
  unsigned short* wqb = xb + nX;
  unsigned short* wkb = wqb + nW;
  unsigned short* wvb = wkb + nW;
  unsigned short* wob = wvb + nW;
  unsigned short* Qb  = wob + nW;
  unsigned short* Kb  = Qb + nX;
  unsigned short* Vb  = Kb + nX;
  unsigned short* ctx = Vb + nX;
  // total: 5*nX + 4*nW ushorts = 88 MiB

  convert_f32_to_bf16<<<4096, 256, 0, stream>>>(x, xb, nX);
  convert_f32_to_bf16<<<1024, 256, 0, stream>>>(wq, wqb, nW);
  convert_f32_to_bf16<<<1024, 256, 0, stream>>>(wk, wkb, nW);
  convert_f32_to_bf16<<<1024, 256, 0, stream>>>(wv, wvb, nW);
  convert_f32_to_bf16<<<1024, 256, 0, stream>>>(wo, wob, nW);

  dim3 gemmGrid(M_ / 128, D_ / 64);  // (64, 16)
  gemm_bf16_wmma<true><<<gemmGrid, 256, 0, stream>>>(xb, wqb, Qb, M_, D_, D_);
  gemm_bf16_wmma<true><<<gemmGrid, 256, 0, stream>>>(xb, wkb, Kb, M_, D_, D_);
  gemm_bf16_wmma<true><<<gemmGrid, 256, 0, stream>>>(xb, wvb, Vb, M_, D_, D_);

  dim3 attnGrid(B_ * H_, S_ / 128);  // (128, 8)
  flash_attn_bf16<<<attnGrid, 256, 0, stream>>>(Qb, Kb, Vb, ctx);

  gemm_bf16_wmma<false><<<gemmGrid, 256, 0, stream>>>(ctx, wob, (float*)d_out,
                                                      M_, D_, D_);
}
